// GradMimicryNetwork_28458453303443
// MI455X (gfx1250) — matrix operand order repaired and checked by reading, not compiled, not run on hardware
//
#include <hip/hip_runtime.h>

// ---------------------------------------------------------------------------
// CDNA5 (gfx1250, wave32) SNN forward: 5x [batched WMMA GEMM + LIF scan]
// ---------------------------------------------------------------------------

typedef __attribute__((ext_vector_type(16))) _Float16 v16h;
typedef __attribute__((ext_vector_type(8)))  _Float16 v8h;
typedef __attribute__((ext_vector_type(8)))  float    v8f;

#define BATCH   512
#define TSTEPS  128
#define K0      784
#define K0P     800   // 784 padded to multiple of 32

// ---------------- weight prep: Wt[n,k] = f16(relu(W[k,n])), zero padded ----
__global__ void prep_weight_kernel(const float* __restrict__ W,
                                   _Float16* __restrict__ Wt,
                                   int K, int N, int Kpad, int Npad)
{
    int idx = blockIdx.x * blockDim.x + threadIdx.x;
    if (idx >= Kpad * Npad) return;
    int n = idx / Kpad;
    int k = idx - n * Kpad;
    float v = 0.0f;
    if (k < K && n < N) {
        v = W[(size_t)k * N + n];
        v = v > 0.0f ? v : 0.0f;       // relu
    }
    Wt[(size_t)n * Kpad + k] = (_Float16)v;
}

// ------------- input transpose: (B,784,T) f32 -> (T,B,800) f16, zero pad ---
__global__ void transpose_input_kernel(const float* __restrict__ in,
                                       _Float16* __restrict__ Xh)
{
    __shared__ float tile[16][17];
    int k0 = blockIdx.x * 16;          // K0P/16 = 50
    int t0 = blockIdx.y * 16;          // T/16   = 8
    int b  = blockIdx.z;               // B      = 512
    int tx = threadIdx.x, ty = threadIdx.y;

    int k = k0 + ty;
    float v = 0.0f;
    if (k < K0) v = in[((size_t)b * K0 + k) * TSTEPS + (t0 + tx)]; // coalesced in t
    tile[ty][tx] = v;
    __syncthreads();
    // Xh[t0+ty][b][k0+tx] = in[b][k0+tx][t0+ty]  (contiguous in k on write)
    Xh[((size_t)(t0 + ty) * BATCH + b) * K0P + (k0 + tx)] = (_Float16)tile[tx][ty];
}

// ------------- batched GEMM: cur[t] = X[t] (BxK) * Wt^T (KxN) --------------
// X : (T, B, Kpad) f16   Wt : (Npad, Kpad) f16   cur : (T, B, Npad) f32
// One wave computes one 16x16 tile of one timestep via v_wmma_f32_16x16x32_f16.
__global__ void wmma_gemm_kernel(const _Float16* __restrict__ X,
                                 const _Float16* __restrict__ Wt,
                                 float* __restrict__ cur,
                                 int Kpad, int Npad)
{
    const int lane = threadIdx.x & 31;
    const int wave = blockIdx.x * (blockDim.x >> 5) + (threadIdx.x >> 5);

    const int mt       = BATCH >> 4;
    const int nt       = Npad >> 4;
    const int tilesPT  = mt * nt;
    if (wave >= tilesPT * TSTEPS) return;          // wave-uniform: EXEC all-1s

    const int t    = wave / tilesPT;
    const int rem  = wave - t * tilesPT;
    const int mti  = rem / nt;
    const int m0   = mti << 4;
    const int n0   = (rem - mti * nt) << 4;

    const int hi   = lane >> 4;                    // lane-half select
    const int l15  = lane & 15;

    // A (16x32 f16): lane row m = l15; halves 0..15 -> K = 32c + hi*16 + i.
    const _Float16* ap = X  + ((size_t)t * BATCH + (m0 + l15)) * Kpad + hi * 16;
    // B (32x16 f16): lane col n = l15; halves 0..15 -> K = 32c + hi*16 + i.
    const _Float16* bp = Wt + (size_t)(n0 + l15) * Kpad + hi * 16;

    v8f acc = {};
    const int chunks = Kpad >> 5;
    for (int c = 0; c < chunks; ++c) {
        v8h  alo = *(const v8h*)(ap);
        v8h  ahi = *(const v8h*)(ap + 8);
        v16h bfrag = *(const v16h*)(bp);
        v16h a;
#pragma unroll
        for (int i = 0; i < 8; ++i) { a[i] = alo[i]; a[8 + i] = ahi[i]; }
        // hint next K-chunk into cache (global_prefetch_b8)
        __builtin_prefetch(ap + 32, 0, 0);
        acc = __builtin_amdgcn_wmma_f32_16x16x32_f16(
                  /*neg_a=*/false, a, /*neg_b=*/false, bfrag,
                  /*c_mod=*/(short)0, acc, /*reuse_a=*/false, /*reuse_b=*/false);
        ap += 32;
        bp += 32;
    }

    // D (16x16 f32): lane col = l15, rows r + 8*hi, r = 0..7
    float* outp = cur + ((size_t)t * BATCH + (m0 + hi * 8)) * Npad + n0 + l15;
#pragma unroll
    for (int r = 0; r < 8; ++r) outp[(size_t)r * Npad] = acc[r];
}

// ------------- LIF scan over time (sequential in t per neuron) -------------
// cur : (T, B, Npad) f32. For shift==1 the drive at t is cur[t-1] (0 at t=0).
// spk_out : (B, N, T) f32 slice of d_out. spk_h : (T, B, N) f16 staging for
// the next layer's GEMM (or null). rates : (B, N) mean firing rate (or null).
__global__ void lif_scan_kernel(const float* __restrict__ cur,
                                float* __restrict__ spk_out,
                                _Float16* __restrict__ spk_h,
                                float* __restrict__ rates,
                                int N, int Npad, int shift)
{
    int idx = blockIdx.x * blockDim.x + threadIdx.x;
    if (idx >= BATCH * N) return;
    int b = idx / N;
    int n = idx - b * N;

    float v = 0.0f;       // V_REST
    float rate = 0.0f;
    for (int t = 0; t < TSTEPS; ++t) {
        float c;
        if (shift) c = (t == 0) ? 0.0f
                                : cur[((size_t)(t - 1) * BATCH + b) * Npad + n];
        else       c = cur[((size_t)t * BATCH + b) * Npad + n];
        v = 0.9f * v + c;                       // decay + integrate
        float s = (v >= 1.0f) ? 1.0f : 0.0f;    // threshold (V_TH = 1)
        spk_out[((size_t)b * N + n) * TSTEPS + t] = s;
        if (spk_h) spk_h[((size_t)t * BATCH + b) * N + n] = (_Float16)s;
        rate += s;
        v = (s > 0.0f) ? 0.0f : v;              // hard reset to V_RESET = 0
    }
    if (rates) rates[(size_t)b * N + n] = rate * (1.0f / (float)TSTEPS);
}

// ---------------------------------------------------------------------------
// Host-side orchestration
// ---------------------------------------------------------------------------
static inline void launch_gemm(const _Float16* X, const _Float16* Wt, float* cur,
                               int Kpad, int Npad, hipStream_t stream)
{
    int waves  = (BATCH >> 4) * (Npad >> 4) * TSTEPS;
    int blocks = (waves + 7) / 8;               // 8 waves (256 threads) / block
    wmma_gemm_kernel<<<blocks, 256, 0, stream>>>(X, Wt, cur, Kpad, Npad);
}

extern "C" void kernel_launch(void* const* d_in, const int* in_sizes, int n_in,
                              void* d_out, int out_size, void* d_ws, size_t ws_size,
                              hipStream_t stream)
{
    (void)in_sizes; (void)n_in; (void)out_size; (void)ws_size;

    const float* in_spikes = (const float*)d_in[0];
    const float* w0 = (const float*)d_in[1];
    const float* w1 = (const float*)d_in[2];
    const float* w2 = (const float*)d_in[3];
    const float* w3 = (const float*)d_in[4];
    const float* w4 = (const float*)d_in[5];
    float* out = (float*)d_out;
    char*  ws  = (char*)d_ws;

    // ---- workspace layout (bytes, 256-aligned) ----
    _Float16* Xh  = (_Float16*)(ws + 0);            // (T,B,800)  104,857,600 B
    _Float16* S0h = (_Float16*)(ws + 104857600);    // (T,B,256)   33,554,432 B
    _Float16* S1h = (_Float16*)(ws + 138412032);    // (T,B,128)   16,777,216 B
    _Float16* S2h = (_Float16*)(ws + 155189248);    // (T,B, 64)    8,388,608 B
    _Float16* S3h = (_Float16*)(ws + 163577856);    // (T,B, 32)    4,194,304 B
    _Float16* W0t = (_Float16*)(ws + 167772160);    // (256,800)      409,600 B
    _Float16* W1t = (_Float16*)(ws + 168181760);    // (128,256)       65,536 B
    _Float16* W2t = (_Float16*)(ws + 168247296);    // ( 64,128)       16,384 B
    _Float16* W3t = (_Float16*)(ws + 168263680);    // ( 32, 64)        4,096 B
    _Float16* W4t = (_Float16*)(ws + 168267776);    // ( 16, 32)        1,024 B
    float*    cur = (float*)   (ws + 168268800);    // (T,B,256) f32 67,108,864 B (reused)

    // ---- output layout (floats, reference return order) ----
    float* o_h0   = out + 0;           // (512,256,128)
    float* o_h1   = out + 16777216;    // (512,128,128)
    float* o_h2   = out + 25165824;    // (512, 64,128)
    float* o_h3   = out + 29360128;    // (512, 32,128)
    float* o_out  = out + 31457280;    // (512, 10,128)
    float* o_rate = out + 32112640;    // (512, 10)

    // ---- prep: relu'd, transposed, f16 weights (K padded to 32, w4 N->16) --
    prep_weight_kernel<<<(800 * 256 + 255) / 256, 256, 0, stream>>>(w0, W0t, 784, 256, 800, 256);
    prep_weight_kernel<<<(256 * 128 + 255) / 256, 256, 0, stream>>>(w1, W1t, 256, 128, 256, 128);
    prep_weight_kernel<<<(128 *  64 + 255) / 256, 256, 0, stream>>>(w2, W2t, 128,  64, 128,  64);
    prep_weight_kernel<<<( 64 *  32 + 255) / 256, 256, 0, stream>>>(w3, W3t,  64,  32,  64,  32);
    prep_weight_kernel<<<( 32 *  16 + 255) / 256, 256, 0, stream>>>(w4, W4t,  32,  10,  32,  16);

    // ---- input spikes -> (T,B,800) f16 ----
    transpose_input_kernel<<<dim3(K0P / 16, TSTEPS / 16, BATCH), dim3(16, 16), 0, stream>>>(in_spikes, Xh);

    // ---- layer 0: driven by input at t (no shift) ----
    launch_gemm(Xh, W0t, cur, 800, 256, stream);
    lif_scan_kernel<<<(BATCH * 256 + 255) / 256, 256, 0, stream>>>(cur, o_h0, S0h, nullptr, 256, 256, 0);

    // ---- layers 1..3: driven by previous-timestep spikes (shift) ----
    launch_gemm(S0h, W1t, cur, 256, 128, stream);
    lif_scan_kernel<<<(BATCH * 128 + 255) / 256, 256, 0, stream>>>(cur, o_h1, S1h, nullptr, 128, 128, 1);

    launch_gemm(S1h, W2t, cur, 128, 64, stream);
    lif_scan_kernel<<<(BATCH * 64 + 255) / 256, 256, 0, stream>>>(cur, o_h2, S2h, nullptr, 64, 64, 1);

    launch_gemm(S2h, W3t, cur, 64, 32, stream);
    lif_scan_kernel<<<(BATCH * 32 + 255) / 256, 256, 0, stream>>>(cur, o_h3, S3h, nullptr, 32, 32, 1);

    // ---- output layer: N=10 padded to 16, emits spikes + firing rates ----
    launch_gemm(S3h, W4t, cur, 32, 16, stream);
    lif_scan_kernel<<<(BATCH * 10 + 255) / 256, 256, 0, stream>>>(cur, o_out, nullptr, o_rate, 10, 16, 1);
}
